// ProductionLiquidNetwork_13795434954878
// MI455X (gfx1250) — matrix operand order repaired and checked by reading, not compiled, not run on hardware
//
#include <hip/hip_runtime.h>
#include <hip/hip_bf16.h>
#include <stddef.h>

// ---------------------------------------------------------------------------
// Problem constants
// ---------------------------------------------------------------------------
#define B_  64
#define T_  1024
#define I_  128
#define H_  512
#define O_  10
#define OPAD 16           // output proj N padded to one WMMA tile
#define TAU_MIN 5.0f
#define TAU_MAX 50.0f

// WMMA fragment vector types (wave32, 16x16x32 bf16)
typedef __attribute__((ext_vector_type(16))) __bf16 v16bf;
typedef __attribute__((ext_vector_type(8)))  __bf16 v8bf;
typedef __attribute__((ext_vector_type(8)))  float  v8f;

__device__ __forceinline__ v8f wmma_bf16(v16bf a, v16bf b, v8f c) {
    // D = A(16x32) * B(32x16) + C(16x16 f32)
    return __builtin_amdgcn_wmma_f32_16x16x32_bf16(
        /*neg_a=*/false, a, /*neg_b=*/false, b,
        /*c_mod=*/(short)0, c, /*reuse_a=*/false, /*reuse_b=*/false);
}

// ---------------------------------------------------------------------------
// B-fragment (packed weights): value(lane, e) laid out contiguously per lane.
// packed[((nt*KT + kt)*32 + lane)*16 + e]  ==  W[K = kt*32 + (lane>>4)*8 + (e&7) + (e>>3)*16]
//                                             [N = nt*16 + (lane&15)]
// -> one 32B contiguous chunk per lane, perfectly coalesced (2x b128 loads).
// ---------------------------------------------------------------------------
__device__ __forceinline__ v16bf load_bfrag(const __bf16* pk, int tileKt, int lane) {
    return *(const v16bf*)(pk + ((size_t)tileKt * 32 + lane) * 16);
}

// ---------------------------------------------------------------------------
// A-fragment from row-major bf16 in LDS (stride in elements, 16B-aligned rows).
// ISA 16-bit A 16x32 layout: lane l (m = l&15, g = l>>4):
//   elems 0..7  -> K = k0 + g*8 .. +7
//   elems 8..15 -> K = k0 + 16 + g*8 .. +7
// ---------------------------------------------------------------------------
__device__ __forceinline__ v16bf load_afrag(const __bf16* A, int stride, int k0, int lane) {
    const int g = lane >> 4, m = lane & 15;
    const __bf16* rp = A + (size_t)m * stride + k0 + g * 8;
    v8bf lo = *(const v8bf*)rp;
    v8bf hi = *(const v8bf*)(rp + 16);
    v16bf a;
#pragma unroll
    for (int i = 0; i < 8; ++i) { a[i] = lo[i]; a[i + 8] = hi[i]; }
    return a;
}

// ---------------------------------------------------------------------------
// Phase 0: pack an f32 weight matrix into the WMMA B-fragment layout (bf16).
// transpose=1 means logical B(k,n) = src[n*ld + k]  (used for W_rec^T).
// Columns >= Nvalid are zero-padded (used for W_out 10 -> 16).
// ---------------------------------------------------------------------------
__global__ void pack_weights_kernel(const float* __restrict__ src,
                                    __bf16* __restrict__ dst,
                                    int K, int Npad, int Nvalid, int ld,
                                    int transpose) {
    const int total = K * Npad;
    int tid = blockIdx.x * blockDim.x + threadIdx.x;
    if (tid >= total) return;
    const int KT = K / 32;
    const int e    = tid & 15;
    const int lane = (tid >> 4) & 31;
    const int rest = tid >> 9;            // nt*KT + kt
    const int kt = rest % KT;
    const int nt = rest / KT;
    const int k = kt * 32 + (lane >> 4) * 8 + (e & 7) + (e >> 3) * 16;
    const int n = nt * 16 + (lane & 15);
    float v = 0.f;
    if (n < Nvalid)
        v = transpose ? src[(size_t)n * ld + k] : src[(size_t)k * ld + n];
    dst[tid] = (__bf16)v;
}

// ---------------------------------------------------------------------------
// Phase 1: xp[t][b][h] = x[b,t,:] @ W_in + b_in   (bf16 out, [T][B][H] layout
// so the scan kernel reads a contiguous 1KB row per batch row per step).
// grid = (T, B/16), block = 256 (8 waves); wave w owns N-tiles w*4..w*4+3.
// ---------------------------------------------------------------------------
__global__ void xp_kernel(const float* __restrict__ x,
                          const float* __restrict__ b_in,
                          const __bf16* __restrict__ pWin,
                          __bf16* __restrict__ xp) {
    const int t  = blockIdx.x;
    const int bt = blockIdx.y;
    const int lane = threadIdx.x & 31;
    const int w    = threadIdx.x >> 5;
    const int g = lane >> 4, nloc = lane & 15, m = lane & 15;

    const float* xrow = x + ((size_t)(bt * 16 + m) * T_ + t) * I_;
    v8f acc[4] = {{}, {}, {}, {}};

#pragma unroll
    for (int kt = 0; kt < I_ / 32; ++kt) {
        const int k0 = kt * 32 + g * 8;
        float4 c0 = *(const float4*)(xrow + k0);
        float4 c1 = *(const float4*)(xrow + k0 + 4);
        float4 c2 = *(const float4*)(xrow + k0 + 16);
        float4 c3 = *(const float4*)(xrow + k0 + 20);
        v16bf a;
        a[0]=(__bf16)c0.x; a[1]=(__bf16)c0.y; a[2]=(__bf16)c0.z; a[3]=(__bf16)c0.w;
        a[4]=(__bf16)c1.x; a[5]=(__bf16)c1.y; a[6]=(__bf16)c1.z; a[7]=(__bf16)c1.w;
        a[8]=(__bf16)c2.x; a[9]=(__bf16)c2.y; a[10]=(__bf16)c2.z; a[11]=(__bf16)c2.w;
        a[12]=(__bf16)c3.x; a[13]=(__bf16)c3.y; a[14]=(__bf16)c3.z; a[15]=(__bf16)c3.w;
#pragma unroll
        for (int j = 0; j < 4; ++j) {
            v16bf b = load_bfrag(pWin, (w * 4 + j) * (I_ / 32) + kt, lane);
            acc[j] = wmma_bf16(a, b, acc[j]);
        }
    }
#pragma unroll
    for (int j = 0; j < 4; ++j) {
        const int col = (w * 4 + j) * 16 + nloc;
        const float bi = b_in[col];
#pragma unroll
        for (int v = 0; v < 8; ++v) {
            const int mrow = v + 8 * g;   // C/D layout: M = vgpr + 8*(lane/16)
            xp[((size_t)t * B_ + bt * 16 + mrow) * H_ + col] = (__bf16)(acc[j][v] + bi);
        }
    }
}

// ---------------------------------------------------------------------------
// Phase 2: persistent scan. 4 workgroups, one per 16-row batch tile, run all
// 1024 timesteps. h state f32 in LDS; ctx=[xp_t | h] bf16 in LDS feeds WMMA;
// weights streamed (bf16, packed) from L2 each step, software-pipelined one
// K-tile ahead in registers so WMMAs never wait on loadcnt==0.
// block = 256 threads = 8 wave32s. Wave w owns H-column tiles w*4..w*4+3 for
// both the recurrent GEMM and tau2 GEMM (so the Euler combine is register-
// local), and tau1 hidden tiles w*2, w*2+1.
// ---------------------------------------------------------------------------
#define CS 1032    // ctx row stride (bf16 elems), 2*H_ + 8 pad
#define HS 516     // h   row stride (f32 elems)
#define DSs 264    // hid row stride (bf16 elems), H_/2 + 8 pad
#define SMEM_BYTES (16*CS*2 + 16*HS*4 + 16*DSs*2 + 16*16*4)

__global__ void scan_kernel(const __bf16* __restrict__ xp,
                            const __bf16* __restrict__ pWrec,
                            const __bf16* __restrict__ pWtau1,
                            const __bf16* __restrict__ pWtau2,
                            const __bf16* __restrict__ pWout,
                            const float* __restrict__ b_tau1,
                            const float* __restrict__ b_tau2,
                            const float* __restrict__ bias,
                            const float* __restrict__ b_out,
                            float* __restrict__ out) {
    extern __shared__ char smem[];
    __bf16* ctx  = (__bf16*)smem;                                  // [16][CS]
    float*  hst  = (float*)(smem + 16 * CS * 2);                   // [16][HS]
    __bf16* hid  = (__bf16*)(smem + 16 * CS * 2 + 16 * HS * 4);    // [16][DSs]
    float*  oacc = (float*)(smem + 16 * CS * 2 + 16 * HS * 4 + 16 * DSs * 2); // [16][16]

    const int tid  = threadIdx.x;
    const int lane = tid & 31;
    const int w    = tid >> 5;
    const int g    = lane >> 4;
    const int nloc = lane & 15;
    const int b0   = blockIdx.x * 16;

    // init state: h = 0 (f32 and bf16 halves)
    for (int i = tid; i < 16 * HS; i += 256) hst[i] = 0.f;
    for (int i = tid; i < 16 * CS; i += 256) ctx[i] = (__bf16)0.f;

    // per-lane column constants (biases) hoisted out of the time loop
    float bias_r[4], b2_r[4], b1_r[2];
#pragma unroll
    for (int j = 0; j < 4; ++j) {
        const int n = (w * 4 + j) * 16 + nloc;
        bias_r[j] = bias[n];
        b2_r[j]   = b_tau2[n];
    }
#pragma unroll
    for (int j = 0; j < 2; ++j) b1_r[j] = b_tau1[(w * 2 + j) * 16 + nloc];
    __syncthreads();

    for (int t = 0; t < T_; ++t) {
        // ---- S0: stage xp_t into ctx x-half; zero output accumulator ------
        {
            const int r = tid >> 4, c = tid & 15;      // 16 rows x 16 chunks of 32 bf16
            const __bf16* srow = xp + ((size_t)t * B_ + b0 + r) * H_ + c * 32;
            __bf16* drow = ctx + r * CS + c * 32;
#pragma unroll
            for (int q = 0; q < 4; ++q)
                *(v8bf*)(drow + q * 8) = *(const v8bf*)(srow + q * 8);
            oacc[tid] = 0.f;
        }
        __syncthreads();

        // ---- S1: rec GEMM (h @ Wrec^T) and tau1 GEMM (ctx @ Wtau1) --------
        // B-tiles double-buffered one K-iteration ahead: WMMA(kt) only waits
        // for loadcnt <= (loads of kt+1), hiding a full K-step of L2 latency.
        v8f accR[4] = {{}, {}, {}, {}};
        {
            v8f acc1[2] = {{}, {}};
            {
                v16bf bR[2][4];
#pragma unroll
                for (int j = 0; j < 4; ++j)
                    bR[0][j] = load_bfrag(pWrec, (w * 4 + j) * (H_ / 32), lane);
#pragma unroll 2
                for (int kt = 0; kt < H_ / 32; ++kt) {             // K = 512
                    const int cur = kt & 1, nxt = cur ^ 1;
                    if (kt + 1 < H_ / 32) {
#pragma unroll
                        for (int j = 0; j < 4; ++j)
                            bR[nxt][j] = load_bfrag(pWrec, (w * 4 + j) * (H_ / 32) + kt + 1, lane);
                    }
                    v16bf a = load_afrag(ctx, CS, H_ + kt * 32, lane); // h half
#pragma unroll
                    for (int j = 0; j < 4; ++j)
                        accR[j] = wmma_bf16(a, bR[cur][j], accR[j]);
                }
            }
            {
                v16bf bT[2][2];
#pragma unroll
                for (int j = 0; j < 2; ++j)
                    bT[0][j] = load_bfrag(pWtau1, (w * 2 + j) * ((2 * H_) / 32), lane);
#pragma unroll 2
                for (int kt = 0; kt < (2 * H_) / 32; ++kt) {       // K = 1024
                    const int cur = kt & 1, nxt = cur ^ 1;
                    if (kt + 1 < (2 * H_) / 32) {
#pragma unroll
                        for (int j = 0; j < 2; ++j)
                            bT[nxt][j] = load_bfrag(pWtau1, (w * 2 + j) * ((2 * H_) / 32) + kt + 1, lane);
                    }
                    v16bf a = load_afrag(ctx, CS, kt * 32, lane);
#pragma unroll
                    for (int j = 0; j < 2; ++j)
                        acc1[j] = wmma_bf16(a, bT[cur][j], acc1[j]);
                }
            }
            // hidden = relu(acc1 + b1) -> LDS (bf16) for tau2 A-matrix
#pragma unroll
            for (int j = 0; j < 2; ++j) {
                const int n = (w * 2 + j) * 16 + nloc;
#pragma unroll
                for (int v = 0; v < 8; ++v) {
                    float hv = acc1[j][v] + b1_r[j];
                    hid[(v + 8 * g) * DSs + n] = (__bf16)(hv > 0.f ? hv : 0.f);
                }
            }
        }
        __syncthreads();

        // ---- S2: tau2 GEMM + fused sigmoid/tanh Euler update --------------
        {
            v8f acc2[4] = {{}, {}, {}, {}};
            {
                v16bf bU[2][4];
#pragma unroll
                for (int j = 0; j < 4; ++j)
                    bU[0][j] = load_bfrag(pWtau2, (w * 4 + j) * ((H_ / 2) / 32), lane);
#pragma unroll 2
                for (int kt = 0; kt < (H_ / 2) / 32; ++kt) {       // K = 256
                    const int cur = kt & 1, nxt = cur ^ 1;
                    if (kt + 1 < (H_ / 2) / 32) {
#pragma unroll
                        for (int j = 0; j < 4; ++j)
                            bU[nxt][j] = load_bfrag(pWtau2, (w * 4 + j) * ((H_ / 2) / 32) + kt + 1, lane);
                    }
                    v16bf a = load_afrag(hid, DSs, kt * 32, lane);
#pragma unroll
                    for (int j = 0; j < 4; ++j)
                        acc2[j] = wmma_bf16(a, bU[cur][j], acc2[j]);
                }
            }
#pragma unroll
            for (int j = 0; j < 4; ++j) {
                const int n = (w * 4 + j) * 16 + nloc;
#pragma unroll
                for (int v = 0; v < 8; ++v) {
                    const int m = v + 8 * g;
                    float z    = acc2[j][v] + b2_r[j];
                    float sig  = 1.f / (1.f + __expf(-z));
                    float tau  = TAU_MIN + sig * (TAU_MAX - TAU_MIN);
                    float xpv  = (float)ctx[m * CS + n];           // xp_t value
                    float hold = hst[m * HS + n];
                    float drv  = tanhf(xpv + accR[j][v] + bias_r[j]);
                    float hnew = hold + (drv - hold) / tau;
                    hst[m * HS + n] = hnew;
                    ctx[m * CS + H_ + n] = (__bf16)hnew;           // next-step operand
                }
            }
        }
        __syncthreads();

        // ---- S3: output projection (K split across 8 waves, ds_add reduce)
        {
            v16bf b0f = load_bfrag(pWout, w * 2 + 0, lane);        // both tiles up front
            v16bf b1f = load_bfrag(pWout, w * 2 + 1, lane);
            v8f acco = {};
            v16bf a0 = load_afrag(ctx, CS, H_ + (w * 2 + 0) * 32, lane); // new_h (bf16)
            v16bf a1 = load_afrag(ctx, CS, H_ + (w * 2 + 1) * 32, lane);
            acco = wmma_bf16(a0, b0f, acco);
            acco = wmma_bf16(a1, b1f, acco);
#pragma unroll
            for (int v = 0; v < 8; ++v)
                atomicAdd(&oacc[(v + 8 * g) * 16 + nloc], acco[v]);
        }
        __syncthreads();

        // ---- S4: emit out[b, t, 0:10] --------------------------------------
        if (tid < 16 * O_) {
            const int m = tid / O_, o = tid % O_;
            out[((size_t)(b0 + m) * T_ + t) * O_ + o] = oacc[m * 16 + o] + b_out[o];
        }
        __syncthreads();
    }
}

// ---------------------------------------------------------------------------
// Workspace layout (bytes)
// ---------------------------------------------------------------------------
#define OFF_WIN   ((size_t)0)                         // 128*512  bf16 = 128KB
#define OFF_WREC  (OFF_WIN  + (size_t)I_ * H_ * 2)    // 512*512  bf16 = 512KB
#define OFF_WT1   (OFF_WREC + (size_t)H_ * H_ * 2)    // 1024*256 bf16 = 512KB
#define OFF_WT2   (OFF_WT1  + (size_t)(2*H_) * (H_/2) * 2) // 256*512 bf16
#define OFF_WOUT  (OFF_WT2  + (size_t)(H_/2) * H_ * 2)     // 512*16 bf16
#define OFF_XP    (OFF_WOUT + (size_t)H_ * OPAD * 2)       // T*B*H bf16 = 64MB

extern "C" void kernel_launch(void* const* d_in, const int* in_sizes, int n_in,
                              void* d_out, int out_size, void* d_ws, size_t ws_size,
                              hipStream_t stream) {
    (void)in_sizes; (void)n_in; (void)out_size; (void)ws_size;
    const float* x      = (const float*)d_in[0];
    const float* W_in   = (const float*)d_in[1];
    const float* b_in   = (const float*)d_in[2];
    const float* W_rec  = (const float*)d_in[3];
    const float* bias   = (const float*)d_in[4];
    const float* W_tau1 = (const float*)d_in[5];
    const float* b_tau1 = (const float*)d_in[6];
    const float* W_tau2 = (const float*)d_in[7];
    const float* b_tau2 = (const float*)d_in[8];
    const float* W_out  = (const float*)d_in[9];
    const float* b_out  = (const float*)d_in[10];
    float* out = (float*)d_out;

    char* ws = (char*)d_ws;
    __bf16* pWin  = (__bf16*)(ws + OFF_WIN);
    __bf16* pWrec = (__bf16*)(ws + OFF_WREC);
    __bf16* pWt1  = (__bf16*)(ws + OFF_WT1);
    __bf16* pWt2  = (__bf16*)(ws + OFF_WT2);
    __bf16* pWout = (__bf16*)(ws + OFF_WOUT);
    __bf16* xp    = (__bf16*)(ws + OFF_XP);

    // Phase 0: pack weights into WMMA B-fragment layout (bf16)
    auto blocks = [](int total) { return (total + 255) / 256; };
    pack_weights_kernel<<<blocks(I_ * H_),        256, 0, stream>>>(W_in,   pWin,  I_,     H_,     H_,     H_,     0);
    pack_weights_kernel<<<blocks(H_ * H_),        256, 0, stream>>>(W_rec,  pWrec, H_,     H_,     H_,     H_,     1); // B = W_rec^T
    pack_weights_kernel<<<blocks(2 * H_ * (H_/2)),256, 0, stream>>>(W_tau1, pWt1,  2 * H_, H_ / 2, H_ / 2, H_ / 2, 0);
    pack_weights_kernel<<<blocks((H_/2) * H_),    256, 0, stream>>>(W_tau2, pWt2,  H_ / 2, H_,     H_,     H_,     0);
    pack_weights_kernel<<<blocks(H_ * OPAD),      256, 0, stream>>>(W_out,  pWout, H_,     OPAD,   O_,     O_,     0);

    // Phase 1: xp = x @ W_in + b_in  (bf16, [T][B][H])
    xp_kernel<<<dim3(T_, B_ / 16), 256, 0, stream>>>(x, b_in, pWin, xp);

    // Phase 2: persistent scan, 4 independent batch-tile chains
    scan_kernel<<<B_ / 16, 256, SMEM_BYTES, stream>>>(
        xp, pWrec, pWt1, pWt2, pWout, b_tau1, b_tau2, bias, b_out, out);
}
